// MultiheadSelfAttentionWithFourier_2113123909966
// MI455X (gfx1250) — matrix operand order repaired and checked by reading, not compiled, not run on hardware
//
#include <hip/hip_runtime.h>
#include <hip/hip_bf16.h>

typedef __attribute__((ext_vector_type(16))) _Float16 v16h;
typedef __attribute__((ext_vector_type(8)))  _Float16 v8h;
typedef __attribute__((ext_vector_type(4)))  _Float16 v4h;
typedef __attribute__((ext_vector_type(8)))  float    v8f;
typedef __attribute__((ext_vector_type(4)))  float    v4f;

#define B_  2
#define S_  2048
#define D_  1024
#define H_  16
#define DK_ 64
#define FP_ 96   // Fourier-augmented head dim (64 + qa + qb + 30 zero pad)

// ---------------- CDNA5 async copy (global -> LDS), with fallbacks ----------
// Builtin signature (from hipcc diagnostic): pointee is GCC-vector int[4],
// global arg in AS1, LDS arg in AS3 (32-bit; low 32 bits of generic = offset).
typedef int gcc_v4i __attribute__((vector_size(16)));
typedef __attribute__((address_space(1))) gcc_v4i* g128p;
typedef __attribute__((address_space(3))) gcc_v4i* l128p;
#define GPTR(x) ((g128p)(size_t)(x))
#define LPTR(x) ((l128p)(unsigned)(size_t)(x))

#if __has_builtin(__builtin_amdgcn_global_load_async_to_lds_b128)
#define HAS_ASYNC 1
#define ASYNC_CP16(g, l) __builtin_amdgcn_global_load_async_to_lds_b128(GPTR(g), LPTR(l), 0, 0)
#else
#define HAS_ASYNC 0
#define ASYNC_CP16(g, l) (*(v8h*)(l) = *(const v8h*)(g))
#endif

#if HAS_ASYNC
#if __has_builtin(__builtin_amdgcn_s_wait_asynccnt)
#define WAIT_ASYNC(n) __builtin_amdgcn_s_wait_asynccnt((unsigned short)(n))
#else
#define WAIT_ASYNC(n) asm volatile("s_wait_asynccnt %0" :: "i"(n) : "memory")
#endif
#else
#define WAIT_ASYNC(n)
#endif

static __device__ __forceinline__ v16h join8(v8h a, v8h b) {
  return __builtin_shufflevector(a, b, 0,1,2,3,4,5,6,7,8,9,10,11,12,13,14,15);
}

// A-fragment (16x32 f16): lane L -> row (L&15); K chunks at (L>>4)*8 and +16
static __device__ __forceinline__ v16h load_a_frag(const _Float16* base, int stride) {
  int lane = threadIdx.x & 31;
  const _Float16* p = base + (size_t)(lane & 15) * stride + ((lane >> 4) << 3);
  v8h lo = *(const v8h*)p;
  v8h hi = *(const v8h*)(p + 16);
  return join8(lo, hi);
}

// B-fragment (32x16 f16): lane L -> column (L&15); K chunks at (L>>4)*8 and +16
static __device__ __forceinline__ v16h load_b_frag(const _Float16* base, int stride) {
  int lane = threadIdx.x & 31;
  const _Float16* p = base + (size_t)(lane & 15) * stride + ((lane >> 4) << 3);
  v8h lo = *(const v8h*)p;
  v8h hi = *(const v8h*)(p + 16);
  return join8(lo, hi);
}

// ---------------------------------------------------------------------------
// f32 -> f16 conversion (one-time pass so GEMM operands are async-copyable)
// ---------------------------------------------------------------------------
__global__ __launch_bounds__(256) void cvt_f16_kernel(
    const float* __restrict__ in, _Float16* __restrict__ out)
{
  size_t i = ((size_t)blockIdx.x * 256 + threadIdx.x) * 4;
  v4f f = *(const v4f*)(in + i);
  *(v4h*)(out + i) = __builtin_convertvector(f, v4h);
}

// ---------------------------------------------------------------------------
// NT GEMM (all-f16 operands): C[M,N] = A[M,K]*B[N,K]^T + bias.
// 128x128 block tile, BK=32, 8 waves (4x2), 32x64 per wave = 2x4 WMMA.
// Double-buffered LDS fed by global_load_async_to_lds_b128 (ASYNCcnt).
// ---------------------------------------------------------------------------
template<bool OUT_F16>
__global__ __launch_bounds__(256) void gemm_nt_f16_kernel(
    const _Float16* __restrict__ A, const _Float16* __restrict__ Bw,
    const float* __restrict__ bias, void* __restrict__ Cv,
    int M, int N, int K)
{
  __shared__ _Float16 As[2][128][40];
  __shared__ _Float16 Bs[2][128][40];
  const int tid  = threadIdx.x;
  const int lane = tid & 31;
  const int w    = tid >> 5;
  const int wm   = (w >> 1) * 32;
  const int wn   = (w & 1) * 64;
  const int tm   = blockIdx.y * 128;
  const int tn   = blockIdx.x * 128;

  // Each thread copies 2 b128 chunks of A and 2 of B per K-step.
  auto issue = [&](int k0, int p) {
    #pragma unroll
    for (int t = 0; t < 2; ++t) {
      int id = tid + t * 256;               // 512 chunks: 128 rows x 4 chunks
      int r = id >> 2, c = (id & 3) << 3;   // c in halves
      ASYNC_CP16(A  + (size_t)(tm + r) * K + k0 + c, &As[p][r][c]);
      ASYNC_CP16(Bw + (size_t)(tn + r) * K + k0 + c, &Bs[p][r][c]);
    }
  };

  const v8f vz = {0.f,0.f,0.f,0.f,0.f,0.f,0.f,0.f};
  v8f acc[2][4];
  #pragma unroll
  for (int i = 0; i < 2; ++i)
    #pragma unroll
    for (int j = 0; j < 4; ++j) acc[i][j] = vz;

  const int NIT = K >> 5;
  issue(0, 0);
  for (int it = 0; it < NIT; ++it) {
    const int p = it & 1;
    const bool more = (it + 1 < NIT);
    if (more) issue((it + 1) << 5, p ^ 1);     // prefetch next tile (shadow buf)
    if (more) { WAIT_ASYNC(4); } else { WAIT_ASYNC(0); }   // retire older batch
    __syncthreads();

    v16h af[2], bf[4];
    #pragma unroll
    for (int i = 0; i < 2; ++i) af[i] = load_a_frag(&As[p][wm + i * 16][0], 40);
    #pragma unroll
    for (int j = 0; j < 4; ++j) bf[j] = load_b_frag(&Bs[p][wn + j * 16][0], 40);
    #pragma unroll
    for (int i = 0; i < 2; ++i)
      #pragma unroll
      for (int j = 0; j < 4; ++j)
        acc[i][j] = __builtin_amdgcn_wmma_f32_16x16x32_f16(
            false, af[i], false, bf[j], (short)0, acc[i][j], false, false);
    __syncthreads();
  }

  #pragma unroll
  for (int i = 0; i < 2; ++i) {
    #pragma unroll
    for (int j = 0; j < 4; ++j) {
      int n = tn + wn + j * 16 + (lane & 15);
      float bv = bias[n];
      #pragma unroll
      for (int r = 0; r < 8; ++r) {
        int m = tm + wm + i * 16 + r + ((lane >> 4) << 3);
        float val = acc[i][j][r] + bv;
        if constexpr (OUT_F16) ((_Float16*)Cv)[(size_t)m * N + n] = (_Float16)val;
        else                   ((float*)Cv)[(size_t)m * N + n]   = val;
      }
    }
  }
}

// ---------------------------------------------------------------------------
// Pack [B,S,D] f16 -> head-major [B*H, S, outw] f16, optional Fourier
// augmentation: out[64]=(sum)/8*scale, out[65]=(alt-sum)/8*scale, pad 0.
// ---------------------------------------------------------------------------
__global__ __launch_bounds__(256) void pack_heads_kernel(
    const _Float16* __restrict__ in, _Float16* __restrict__ out,
    int outw, float scale, int aug)
{
  int g    = blockIdx.x * 8 + (threadIdx.x >> 5);   // row in [0, B*S*H)
  int lane = threadIdx.x & 31;
  int b    = g / (S_ * H_);
  int rem  = g - b * (S_ * H_);
  int s    = rem / H_;
  int h    = rem - s * H_;
  const _Float16* ip = in + ((size_t)(b * S_ + s)) * D_ + h * DK_;
  float e0 = (float)ip[lane];
  float e1 = (float)ip[lane + 32];
  _Float16* op = out + ((size_t)((b * H_ + h) * S_ + s)) * outw;
  op[lane]      = (_Float16)(e0 * scale);
  op[lane + 32] = (_Float16)(e1 * scale);
  if (aug) {
    float ps = e0 + e1;                       // (-1)^d identical for d, d+32
    float pa = (lane & 1) ? -ps : ps;
    #pragma unroll
    for (int m = 1; m < 32; m <<= 1) {
      ps += __shfl_xor(ps, m, 32);
      pa += __shfl_xor(pa, m, 32);
    }
    float v = 0.f;
    if (lane == 0) v = ps * 0.125f * scale;   // rfft bin 0
    if (lane == 1) v = pa * 0.125f * scale;   // rfft bin 32 (Nyquist)
    op[64 + lane] = (_Float16)v;              // lanes 2..31 write zero pad
  }
}

// ---------------------------------------------------------------------------
// FlashAttention over augmented logits. Grid: (S/128, B*H). 8 waves/block,
// each wave owns a 16-query stripe + full online-softmax state.
// Logits pre-scaled by (1/16)*log2(e) via Q packing -> exp2 softmax.
// ---------------------------------------------------------------------------
__global__ __launch_bounds__(256) void attn_kernel(
    const _Float16* __restrict__ Qp, const _Float16* __restrict__ Kp,
    const _Float16* __restrict__ Vp, _Float16* __restrict__ Op)
{
  __shared__ _Float16 Ks[64][104];     // [key][feat 0..95], padded stride
  __shared__ _Float16 Vt[64][72];      // [dcol][key], padded stride
  __shared__ _Float16 Ps[8][16][40];   // per-wave P staging 16x32, padded

  const int tid  = threadIdx.x, lane = tid & 31, w = tid >> 5;
  const int bh   = blockIdx.y;
  const int b    = bh >> 4, h = bh & 15;
  const int qbase = blockIdx.x * 128 + w * 16;

  const _Float16* qrow =
      Qp + ((size_t)bh * S_ + qbase + (lane & 15)) * FP_ + ((lane >> 4) << 3);
  v16h qf[3];
  #pragma unroll
  for (int kk = 0; kk < 3; ++kk) {
    v8h lo = *(const v8h*)(qrow + kk * 32);
    v8h hi = *(const v8h*)(qrow + kk * 32 + 16);
    qf[kk] = join8(lo, hi);
  }

  const v8f vz = {0.f,0.f,0.f,0.f,0.f,0.f,0.f,0.f};
  v8f o[4];
  #pragma unroll
  for (int j = 0; j < 4; ++j) o[j] = vz;
  float mrow[8], lrow[8];
  #pragma unroll
  for (int r = 0; r < 8; ++r) { mrow[r] = -1e30f; lrow[r] = 0.f; }

  for (int kv0 = 0; kv0 < S_; kv0 += 64) {
    #pragma unroll
    for (int t = 0; t < 3; ++t) {                 // K block 64x96 halves (async)
      int idx = tid + t * 256;
      int r = idx / 12, c = (idx % 12) << 3;
      ASYNC_CP16(Kp + ((size_t)bh * S_ + kv0 + r) * FP_ + c, &Ks[r][c]);
    }
    #pragma unroll
    for (int t = 0; t < 2; ++t) {                 // V block 64x64, transpose
      int idx = tid + t * 256;
      int r = idx >> 3, c = (idx & 7) << 3;
      v8h vv = *(const v8h*)(Vp + ((size_t)bh * S_ + kv0 + r) * DK_ + c);
      #pragma unroll
      for (int e = 0; e < 8; ++e) Vt[c + e][r] = vv[e];
    }
    WAIT_ASYNC(0);
    __syncthreads();

    #pragma unroll
    for (int cc = 0; cc < 2; ++cc) {              // 32-key chunks
      v8f s0 = vz, s1 = vz;
      #pragma unroll
      for (int kk = 0; kk < 3; ++kk) {            // K-dim 96 = 3x32
        v16h k0f = load_b_frag(&Ks[cc * 32 +  0][kk * 32], 104);
        v16h k1f = load_b_frag(&Ks[cc * 32 + 16][kk * 32], 104);
        s0 = __builtin_amdgcn_wmma_f32_16x16x32_f16(false, qf[kk], false, k0f, (short)0, s0, false, false);
        s1 = __builtin_amdgcn_wmma_f32_16x16x32_f16(false, qf[kk], false, k1f, (short)0, s1, false, false);
      }
      float al[8];
      #pragma unroll
      for (int r = 0; r < 8; ++r) {
        float cm = fmaxf(s0[r], s1[r]);
        #pragma unroll
        for (int m = 1; m < 16; m <<= 1) cm = fmaxf(cm, __shfl_xor(cm, m, 32));
        float nm = fmaxf(mrow[r], cm);
        al[r] = exp2f(mrow[r] - nm);
        mrow[r] = nm;
        float p0 = exp2f(s0[r] - nm);
        float p1 = exp2f(s1[r] - nm);
        s0[r] = p0; s1[r] = p1;
        float rs = p0 + p1;
        #pragma unroll
        for (int m = 1; m < 16; m <<= 1) rs += __shfl_xor(rs, m, 32);
        lrow[r] = lrow[r] * al[r] + rs;
      }
      #pragma unroll
      for (int j = 0; j < 4; ++j)
        #pragma unroll
        for (int r = 0; r < 8; ++r) o[j][r] *= al[r];

      #pragma unroll
      for (int r = 0; r < 8; ++r) {               // P: C layout -> LDS
        int row = r + ((lane >> 4) << 3);
        Ps[w][row][(lane & 15)]      = (_Float16)s0[r];
        Ps[w][row][(lane & 15) + 16] = (_Float16)s1[r];
      }
      v16h pf = load_a_frag(&Ps[w][0][0], 40);    // -> A fragment (in-order LDS)
      #pragma unroll
      for (int j = 0; j < 4; ++j) {
        v16h vf = load_b_frag(&Vt[j * 16][cc * 32], 72);
        o[j] = __builtin_amdgcn_wmma_f32_16x16x32_f16(false, pf, false, vf, (short)0, o[j], false, false);
      }
    }
    __syncthreads();
  }

  float inv[8];
  #pragma unroll
  for (int r = 0; r < 8; ++r) inv[r] = 1.0f / lrow[r];
  #pragma unroll
  for (int j = 0; j < 4; ++j) {
    int n = h * DK_ + j * 16 + (lane & 15);
    #pragma unroll
    for (int r = 0; r < 8; ++r) {
      int srow = qbase + r + ((lane >> 4) << 3);
      Op[((size_t)(b * S_ + srow)) * D_ + n] = (_Float16)(o[j][r] * inv[r]);
    }
  }
}

// ---------------------------------------------------------------------------
extern "C" void kernel_launch(void* const* d_in, const int* in_sizes, int n_in,
                              void* d_out, int out_size, void* d_ws, size_t ws_size,
                              hipStream_t stream)
{
  (void)in_sizes; (void)n_in; (void)out_size; (void)ws_size;
  const float* x  = (const float*)d_in[0];
  const float* Wq = (const float*)d_in[1];
  const float* bq = (const float*)d_in[2];
  const float* Wk = (const float*)d_in[3];
  const float* bk = (const float*)d_in[4];
  const float* Wv = (const float*)d_in[5];
  const float* bv = (const float*)d_in[6];
  const float* Wo = (const float*)d_in[7];
  const float* bo = (const float*)d_in[8];

  char* ws = (char*)d_ws;
  const size_t MB = 1u << 20;
  _Float16* xh  = (_Float16*)(ws);               //  8 MB
  _Float16* Wqh = (_Float16*)(ws +  8 * MB);     //  2 MB
  _Float16* Wkh = (_Float16*)(ws + 10 * MB);     //  2 MB
  _Float16* Wvh = (_Float16*)(ws + 12 * MB);     //  2 MB
  _Float16* Woh = (_Float16*)(ws + 14 * MB);     //  2 MB
  _Float16* Qt  = (_Float16*)(ws + 16 * MB);     //  8 MB
  _Float16* Kt  = (_Float16*)(ws + 24 * MB);     //  8 MB
  _Float16* Vtm = (_Float16*)(ws + 32 * MB);     //  8 MB
  _Float16* Qp  = (_Float16*)(ws + 40 * MB);     // 12 MB
  _Float16* Kp  = (_Float16*)(ws + 52 * MB);     // 12 MB
  _Float16* Vp  = (_Float16*)(ws + 64 * MB);     //  8 MB
  _Float16* Op  = (_Float16*)(ws + 72 * MB);     //  8 MB  (end: 80 MB)

  // one-time f32 -> f16 operand conversion
  cvt_f16_kernel<<<dim3(4096), 256, 0, stream>>>(x,  xh);
  cvt_f16_kernel<<<dim3(1024), 256, 0, stream>>>(Wq, Wqh);
  cvt_f16_kernel<<<dim3(1024), 256, 0, stream>>>(Wk, Wkh);
  cvt_f16_kernel<<<dim3(1024), 256, 0, stream>>>(Wv, Wvh);
  cvt_f16_kernel<<<dim3(1024), 256, 0, stream>>>(Wo, Woh);

  dim3 gg(8, 32), gb(256);
  gemm_nt_f16_kernel<true ><<<gg, gb, 0, stream>>>(xh, Wqh, bq, Qt,  4096, 1024, 1024);
  gemm_nt_f16_kernel<true ><<<gg, gb, 0, stream>>>(xh, Wkh, bk, Kt,  4096, 1024, 1024);
  gemm_nt_f16_kernel<true ><<<gg, gb, 0, stream>>>(xh, Wvh, bv, Vtm, 4096, 1024, 1024);

  // logits = (q.k + qa*ka + qb*kb)/16 ; fold 1/16 and log2(e) into Q
  const float QSC = (1.0f / 16.0f) * 1.44269504088896340736f;
  dim3 pg((B_ * S_ * H_) / 8), pb(256);
  pack_heads_kernel<<<pg, pb, 0, stream>>>(Qt,  Qp, 96, QSC, 1);
  pack_heads_kernel<<<pg, pb, 0, stream>>>(Kt,  Kp, 96, 1.0f, 1);
  pack_heads_kernel<<<pg, pb, 0, stream>>>(Vtm, Vp, 64, 1.0f, 0);

  attn_kernel<<<dim3(16, 32), 256, 0, stream>>>(Qp, Kp, Vp, Op);

  gemm_nt_f16_kernel<false><<<gg, gb, 0, stream>>>(Op, Woh, bo, (float*)d_out, 4096, 1024, 1024);
}
